// RBF_gaussian_16389595201656
// MI455X (gfx1250) — matrix operand order repaired and checked by reading, not compiled, not run on hardware
//
#include <hip/hip_runtime.h>
#include <hip/hip_bf16.h>

// RBF gaussian: out[b,o] = exp(-(||x_b||^2 - 2 x_b.c_o + ||c_o||^2) * exp(-2 log_sigma_o))
// B=65536, OUT=2048, IN=64.  Output-store bound (~537MB); GEMM via f16 WMMA, fp32 accum.

#define RBF_B   65536
#define RBF_OUT 2048
#define RBF_IN  64

typedef __attribute__((ext_vector_type(16))) _Float16 v16h;
typedef __attribute__((ext_vector_type(8)))  _Float16 v8h;
typedef __attribute__((ext_vector_type(8)))  float    v8f;

#define LOG2E 1.4426950408889634f

// Load one 16-element A fragment slice (16-bit A 16x32 layout): lane group g takes
// K-chunks [base, base+8) and [base+16, base+24); caller passes p = row + base.
// Accumulates fp32 sum-of-squares of the 16 loaded values into ss.
static __device__ __forceinline__ v16h load_a_frag(const float* __restrict__ p, float& ss) {
    float4 q0 = *(const float4*)(p + 0);
    float4 q1 = *(const float4*)(p + 4);
    float4 q2 = *(const float4*)(p + 16);
    float4 q3 = *(const float4*)(p + 20);
    ss += q0.x*q0.x + q0.y*q0.y + q0.z*q0.z + q0.w*q0.w;
    ss += q1.x*q1.x + q1.y*q1.y + q1.z*q1.z + q1.w*q1.w;
    ss += q2.x*q2.x + q2.y*q2.y + q2.z*q2.z + q2.w*q2.w;
    ss += q3.x*q3.x + q3.y*q3.y + q3.z*q3.z + q3.w*q3.w;
    v16h r;
    r[0]=(_Float16)q0.x;  r[1]=(_Float16)q0.y;  r[2]=(_Float16)q0.z;  r[3]=(_Float16)q0.w;
    r[4]=(_Float16)q1.x;  r[5]=(_Float16)q1.y;  r[6]=(_Float16)q1.z;  r[7]=(_Float16)q1.w;
    r[8]=(_Float16)q2.x;  r[9]=(_Float16)q2.y;  r[10]=(_Float16)q2.z; r[11]=(_Float16)q2.w;
    r[12]=(_Float16)q3.x; r[13]=(_Float16)q3.y; r[14]=(_Float16)q3.z; r[15]=(_Float16)q3.w;
    return r;
}

// Load one 16-element B fragment slice (16-bit B layout: per lane, K-chunks
// [base, base+8) and [base+16, base+24)). p = row + base: two b128 loads.
static __device__ __forceinline__ v16h load_b_frag(const _Float16* __restrict__ p) {
    v8h lo = *(const v8h*)(p);
    v8h hi = *(const v8h*)(p + 16);
    v16h r;
#pragma unroll
    for (int i = 0; i < 8; ++i) { r[i] = lo[i]; r[i + 8] = hi[i]; }
    return r;
}

// ---- Prep: centers fp32 -> f16 (row-major), c2[o] = sum(c^2) fp32, isg2[o] = exp(-2*ls)
__global__ __launch_bounds__(32) void rbf_prep_kernel(
    const float* __restrict__ centers, const float* __restrict__ log_sigmas,
    _Float16* __restrict__ cf16, float* __restrict__ c2, float* __restrict__ isg2)
{
    const int o = blockIdx.x;
    const int t = threadIdx.x;           // 0..31, wave32
    const float* row = centers + (size_t)o * RBF_IN;
    float a = row[t];
    float b = row[t + 32];
    cf16[(size_t)o * RBF_IN + t]      = (_Float16)a;
    cf16[(size_t)o * RBF_IN + t + 32] = (_Float16)b;
    float ss = a * a + b * b;
#pragma unroll
    for (int off = 16; off > 0; off >>= 1)
        ss += __shfl_xor(ss, off, 32);
    if (t == 0) {
        c2[o]   = ss;
        isg2[o] = __builtin_amdgcn_exp2f(-2.0f * log_sigmas[o] * LOG2E);
    }
}

// ---- Main: 8 waves/block, each wave owns a 16-row M tile, sweeps all 128 N tiles.
__global__ __launch_bounds__(256) void rbf_gaussian_wmma_kernel(
    const float* __restrict__ x,
    const _Float16* __restrict__ cf16,
    const float* __restrict__ c2,
    const float* __restrict__ isg2,
    float* __restrict__ out)
{
    const int lane = threadIdx.x & 31;
    const int wave = threadIdx.x >> 5;
    const int g    = lane >> 4;      // lane half-group (K-chunk / M-high selector)
    const int rIn  = lane & 15;      // A/B source row within tile == output column N
    const int mBase = (blockIdx.x * 8 + wave) * 16;

    __shared__ float x2s[8][16];

    // A fragments: x rows (fp32 -> f16), K = 0..31 and 32..63; fp32 ||x||^2 on the fly.
    const float* xrow = x + (size_t)(mBase + rIn) * RBF_IN;
    float ss = 0.0f;
    v16h a0 = load_a_frag(xrow + g * 8,      ss);
    v16h a1 = load_a_frag(xrow + 32 + g * 8, ss);
    // lane l and l^16 together cover the full row -> full-row sumsq everywhere
    float x2 = ss + __shfl_xor(ss, 16, 32);
    if (lane < 16) x2s[wave][lane] = x2;
    __syncthreads();
    float x2v[8];
#pragma unroll
    for (int r = 0; r < 8; ++r) x2v[r] = x2s[wave][g * 8 + r];   // C/D rows this lane owns

    const int ntiles = RBF_OUT / 16;   // 128
    for (int t = 0; t < ntiles; ++t) {
        const int col = t * 16 + rIn;  // centers row for B load == output column
        const _Float16* bp = cf16 + (size_t)col * RBF_IN + g * 8;
        v16h b0 = load_b_frag(bp);        // K  0..31 slice
        v16h b1 = load_b_frag(bp + 32);   // K 32..63 slice
        if (t + 1 < ntiles)
            __builtin_prefetch(cf16 + (size_t)(col + 16) * RBF_IN, 0, 0);

        v8f acc = {};
        acc = __builtin_amdgcn_wmma_f32_16x16x32_f16(false, a0, false, b0,
                                                     (short)0, acc, false, false);
        acc = __builtin_amdgcn_wmma_f32_16x16x32_f16(false, a1, false, b1,
                                                     (short)0, acc, false, false);

        const float c2v = c2[col];
        const float is2 = isg2[col];
        float* orow = out + (size_t)(mBase + g * 8) * RBF_OUT + col;
#pragma unroll
        for (int r = 0; r < 8; ++r) {
            float sq = x2v[r] - 2.0f * acc[r] + c2v;
            sq = fmaxf(sq, 0.0f);
            orow[(size_t)r * RBF_OUT] = __builtin_amdgcn_exp2f(-(sq * is2) * LOG2E);
        }
    }
}

extern "C" void kernel_launch(void* const* d_in, const int* in_sizes, int n_in,
                              void* d_out, int out_size, void* d_ws, size_t ws_size,
                              hipStream_t stream) {
    const float* x       = (const float*)d_in[0];   // [65536, 64]
    const float* centers = (const float*)d_in[1];   // [2048, 64]
    const float* ls      = (const float*)d_in[2];   // [2048]
    float* out = (float*)d_out;                     // [65536, 2048]

    char* ws = (char*)d_ws;
    _Float16* cf16 = (_Float16*)ws;                                   // 2048*64*2 = 262144 B
    float*    c2   = (float*)(ws + (size_t)RBF_OUT * RBF_IN * 2);     // 8192 B
    float*    isg2 = (float*)(ws + (size_t)RBF_OUT * RBF_IN * 2 + RBF_OUT * 4);

    rbf_prep_kernel<<<RBF_OUT, 32, 0, stream>>>(centers, ls, cf16, c2, isg2);
    rbf_gaussian_wmma_kernel<<<RBF_B / 128, 256, 0, stream>>>(x, cf16, c2, isg2, out);
}
